// Graph_drug_55353538511280
// MI455X (gfx1250) — matrix operand order repaired and checked by reading, not compiled, not run on hardware
//
#include <hip/hip_runtime.h>
#include <hip/hip_bf16.h>

// ---------------------------------------------------------------------------
// GraphSAGE (2 stacks x 3 layers) + global mean pool, bf16 WMMA GEMMs.
//
// Per layer:  out = ReLU( agg @ Wl^T + bl + x @ Wr^T )
//   1) scatter:  agg_f32[dst] += x_bf16[src]      (global_atomic_add_f32)
//   2) convert:  agg_f32 -> agg_bf16 (row stride padded to mult of 32, zeros)
//   3) fused GEMM: two v_wmma_f32_16x16x32_bf16 per K-step (agg*Wl + x*Wr),
//      bias + ReLU epilogue, bf16 output with padded stride (input of next).
// Pool: f32 atomics into [G,128] sums + counts, then finalize divide+concat.
// ---------------------------------------------------------------------------

typedef __attribute__((ext_vector_type(16))) __bf16 v16bf;
typedef __attribute__((ext_vector_type(8)))  __bf16 v8bf;
typedef __attribute__((ext_vector_type(8)))  float  v8f;

#define NUM_GRAPHS 4096
#define POOL_D     128

static inline size_t r32(size_t x) { return (x + 31) & ~(size_t)31; }
static inline size_t r16(size_t x) { return (x + 15) & ~(size_t)15; }
static inline unsigned cdiv(size_t a, size_t b) { return (unsigned)((a + b - 1) / b); }

// --------------------------- kernels ---------------------------------------

// f32 [R x K] (row-major, unpadded) -> bf16 [Rpad? x Kpad], zero padding.
// total = Rout * Kpad elements; rows >= R and cols >= K get 0.
__global__ void k_cvt_pad(const float* __restrict__ src, int R, int K,
                          __bf16* __restrict__ dst, int Kpad, long total) {
  long t = (long)blockIdx.x * blockDim.x + threadIdx.x;
  if (t >= total) return;
  long r = t / Kpad;
  int  c = (int)(t % Kpad);
  float v = (r < (long)R && c < K) ? src[r * (long)K + c] : 0.0f;
  dst[t] = (__bf16)v;
}

// Edge scatter-add: agg[dst] += X[src], X bf16 (stride ldx), agg f32 (stride K).
// One thread handles 4 feature dims of one edge.
__global__ void k_edge_scatter(const __bf16* __restrict__ X, int ldx,
                               const int* __restrict__ ei, int E,
                               float* __restrict__ agg, int K) {
  int chunks = (K + 3) >> 2;
  long t = (long)blockIdx.x * blockDim.x + threadIdx.x;
  long total = (long)E * chunks;
  if (t >= total) return;
  int e  = (int)(t / chunks);
  int d0 = (int)(t % chunks) * 4;
  int src = ei[e];
  int dst = ei[E + e];
  const __bf16* xp = X + (size_t)src * ldx;
  float* ap = agg + (size_t)dst * K;
#pragma unroll
  for (int j = 0; j < 4; ++j) {
    int d = d0 + j;
    if (d < K) atomicAdd(ap + d, (float)xp[d]);
  }
}

// A-fragment loader: 16x32 bf16, ISA layout:
//   lane (m = l&15, h = l>>4): elems 0..7  = row[kb + 8h + 0..7]
//                              elems 8..15 = row[kb + 16 + 8h + 0..7]
__device__ __forceinline__ v16bf load_a_frag(const __bf16* base, int lda,
                                             int row, int kb, int h) {
  const __bf16* p = base + (size_t)row * lda + kb;
  v8bf lo = *reinterpret_cast<const v8bf*>(p + 8 * h);
  v8bf hi = *reinterpret_cast<const v8bf*>(p + 16 + 8 * h);
  v16bf r;
#pragma unroll
  for (int i = 0; i < 8; ++i) { r[i] = lo[i]; r[i + 8] = hi[i]; }
  return r;
}

// B-fragment loader: 32x16 bf16 (= W^T), lane (n = l&15, h = l>>4):
//   elems 0..7 = W[n][kb + 8h + 0..7], elems 8..15 = W[n][kb + 16 + 8h + 0..7]
__device__ __forceinline__ v16bf load_b_frag(const __bf16* w, int ldb,
                                             int wrow, int kb, int h) {
  const __bf16* p = w + (size_t)wrow * ldb + kb;
  v8bf lo = *reinterpret_cast<const v8bf*>(p + 8 * h);
  v8bf hi = *reinterpret_cast<const v8bf*>(p + 16 + 8 * h);
  v16bf r;
#pragma unroll
  for (int i = 0; i < 8; ++i) { r[i] = lo[i]; r[i + 8] = hi[i]; }
  return r;
}

// Fused SAGE GEMM: C = ReLU(A1 @ B1^T + A2 @ B2^T + bias), bf16 in/out, f32 acc.
// A1/A2: [16*nRowTiles x Kpad] bf16 (stride lda). B1/B2: [Opad x Kpad] bf16.
// Each wave computes one 16x16 output tile; 8 waves (8 row tiles) per block.
__global__ void __launch_bounds__(256)
k_sage_gemm(const __bf16* __restrict__ A1, const __bf16* __restrict__ A2, int lda,
            const __bf16* __restrict__ B1, const __bf16* __restrict__ B2, int ldb,
            const float* __restrict__ bias, __bf16* __restrict__ C, int ldc,
            int nRowTiles, int Kpad, int O) {
  int lane = threadIdx.x & 31;
  int wave = threadIdx.x >> 5;
  int rt = blockIdx.y * 8 + wave;          // row tile (wave-uniform)
  if (rt >= nRowTiles) return;             // uniform exit: EXEC stays all-ones
  int ct = blockIdx.x;                     // col tile
  int h  = lane >> 4;
  int mn = lane & 15;
  int arow = rt * 16 + mn;                 // activation row for A fragments
  int wrow = ct * 16 + mn;                 // weight row (output column)

  v8f acc = {};
  for (int kb = 0; kb < Kpad; kb += 32) {
    v16bf a1 = load_a_frag(A1, lda, arow, kb, h);
    v16bf b1 = load_b_frag(B1, ldb, wrow, kb, h);
    acc = __builtin_amdgcn_wmma_f32_16x16x32_bf16(false, a1, false, b1,
                                                  (short)0, acc, false, false);
    v16bf a2 = load_a_frag(A2, lda, arow, kb, h);
    v16bf b2 = load_b_frag(B2, ldb, wrow, kb, h);
    acc = __builtin_amdgcn_wmma_f32_16x16x32_bf16(false, a2, false, b2,
                                                  (short)0, acc, false, false);
  }

  int col = ct * 16 + mn;                  // output column for this lane
  if (col < O) {
    float bv = bias[col];
#pragma unroll
    for (int v = 0; v < 8; ++v) {
      int m = v + 8 * h;                   // C/D layout: vgpr v, lane-half h
      float val = acc[v] + bv;
      val = val > 0.0f ? val : 0.0f;       // ReLU
      C[(size_t)(rt * 16 + m) * ldc + col] = (__bf16)val;
    }
  }
}

// Mean-pool accumulate: 32 threads per node, each handles 4 of 128 dims.
__global__ void k_pool(const __bf16* __restrict__ H, int ldh,
                       const int* __restrict__ batch, int N,
                       float* __restrict__ sums, float* __restrict__ cnt) {
  long t = (long)blockIdx.x * blockDim.x + threadIdx.x;
  int n = (int)(t >> 5);
  int q = (int)(t & 31);
  if (n >= N) return;
  int g = batch[n];
  const __bf16* hp = H + (size_t)n * ldh + q * 4;
  float* sp = sums + (size_t)g * POOL_D + q * 4;
#pragma unroll
  for (int j = 0; j < 4; ++j) atomicAdd(sp + j, (float)hp[j]);
  if (q == 0) atomicAdd(cnt + g, 1.0f);
}

// out[g, 0:128] = sums_m/cnt_m ; out[g, 128:256] = sums_c/cnt_c
__global__ void k_finalize(const float* __restrict__ sm, const float* __restrict__ cm,
                           const float* __restrict__ sc, const float* __restrict__ cc,
                           float* __restrict__ out) {
  int t = blockIdx.x * blockDim.x + threadIdx.x;
  if (t >= NUM_GRAPHS * 256) return;
  int g = t >> 8;
  int d = t & 255;
  float v;
  if (d < POOL_D) v = sm[(size_t)g * POOL_D + d] / fmaxf(cm[g], 1.0f);
  else            v = sc[(size_t)g * POOL_D + (d - POOL_D)] / fmaxf(cc[g], 1.0f);
  out[t] = v;
}

// --------------------------- host driver ------------------------------------

struct LayerW { const float *wl, *bl, *wr; int O; };

static void run_stack(const float* x, int N, int K0,
                      const int* ei, int E, const int* batch,
                      const LayerW L[3], __bf16* const wb[6],
                      __bf16* inb, __bf16* outb, float* aggf, __bf16* aggb,
                      float* sums, float* cnt, hipStream_t s) {
  const int B = 256;
  int Kin = K0;
  int sIn = (int)r32(K0);
  {                                         // x (f32) -> inb (bf16 padded)
    long total = (long)N * sIn;
    k_cvt_pad<<<cdiv(total, B), B, 0, s>>>(x, N, Kin, inb, sIn, total);
  }
  __bf16* cur = inb;
  __bf16* nxt = outb;
  for (int i = 0; i < 3; ++i) {
    int Oi = L[i].O;
    int sOut = (int)r32(Oi);
    hipMemsetAsync(aggf, 0, (size_t)N * Kin * sizeof(float), s);
    {
      int chunks = (Kin + 3) >> 2;
      long total = (long)E * chunks;
      k_edge_scatter<<<cdiv(total, B), B, 0, s>>>(cur, sIn, ei, E, aggf, Kin);
    }
    {
      long total = (long)N * sIn;
      k_cvt_pad<<<cdiv(total, B), B, 0, s>>>(aggf, N, Kin, aggb, sIn, total);
    }
    hipMemsetAsync(nxt, 0, (size_t)N * sOut * sizeof(__bf16), s);
    {
      int nRT = N / 16;                    // N divisible by 16 for both stacks
      dim3 grid(cdiv(Oi, 16), cdiv(nRT, 8));
      k_sage_gemm<<<grid, 256, 0, s>>>(aggb, cur, sIn, wb[2 * i], wb[2 * i + 1],
                                       sIn, L[i].bl, nxt, sOut, nRT, sIn, Oi);
    }
    __bf16* t = cur; cur = nxt; nxt = t;
    Kin = Oi; sIn = sOut;
  }
  // cur = h3 bf16, stride 128
  hipMemsetAsync(sums, 0, (size_t)NUM_GRAPHS * POOL_D * sizeof(float), s);
  hipMemsetAsync(cnt, 0, (size_t)NUM_GRAPHS * sizeof(float), s);
  {
    long total = (long)N * 32;
    k_pool<<<cdiv(total, B), B, 0, s>>>(cur, POOL_D, batch, N, sums, cnt);
  }
}

extern "C" void kernel_launch(void* const* d_in, const int* in_sizes, int n_in,
                              void* d_out, int out_size, void* d_ws, size_t ws_size,
                              hipStream_t stream) {
  (void)in_sizes; (void)n_in; (void)out_size; (void)ws_size;
  const int N_MOL = 200000, E_MOL = 1000000;
  const int N_CLQ = 120000, E_CLQ = 500000;

  const float* x      = (const float*)d_in[0];
  const int*   ei     = (const int*)  d_in[1];
  const int*   batch  = (const int*)  d_in[2];
  const float* xc     = (const float*)d_in[3];
  const int*   eic    = (const int*)  d_in[4];
  const int*   batchc = (const int*)  d_in[5];

  LayerW Lm[3] = {
      {(const float*)d_in[6],  (const float*)d_in[7],  (const float*)d_in[8],  156},
      {(const float*)d_in[9],  (const float*)d_in[10], (const float*)d_in[11], 312},
      {(const float*)d_in[12], (const float*)d_in[13], (const float*)d_in[14], 128}};
  LayerW Lc[3] = {
      {(const float*)d_in[15], (const float*)d_in[16], (const float*)d_in[17], 184},
      {(const float*)d_in[18], (const float*)d_in[19], (const float*)d_in[20], 368},
      {(const float*)d_in[21], (const float*)d_in[22], (const float*)d_in[23], 128}};
  const int Km[3] = {78, 156, 312};
  const int Kc[3] = {92, 184, 368};

  // ---- carve workspace (deterministic offsets, 256B aligned) ----
  char* base = (char*)d_ws;
  size_t off = 0;
  auto alloc = [&](size_t bytes) -> char* {
    char* r = base + off;
    off = (off + bytes + 255) & ~(size_t)255;
    return r;
  };
  size_t actBytes = (size_t)N_MOL * 320 * 2;     // >= N_CLQ*384*2
  __bf16* inb  = (__bf16*)alloc(actBytes);
  __bf16* outb = (__bf16*)alloc(actBytes);
  float*  aggf = (float*) alloc((size_t)N_MOL * 312 * 4);  // >= N_CLQ*368*4
  __bf16* aggb = (__bf16*)alloc(actBytes);
  float* sums_m = (float*)alloc((size_t)NUM_GRAPHS * POOL_D * 4);
  float* cnt_m  = (float*)alloc((size_t)NUM_GRAPHS * 4);
  float* sums_c = (float*)alloc((size_t)NUM_GRAPHS * POOL_D * 4);
  float* cnt_c  = (float*)alloc((size_t)NUM_GRAPHS * 4);

  // pre-converted bf16 weights: rows padded to 16, cols to 32, zero filled
  __bf16* wbm[6];
  __bf16* wbc[6];
  auto cvt_w = [&](const float* w, int O, int K) -> __bf16* {
    int Opad = (int)r16(O), Kpad = (int)r32(K);
    __bf16* dst = (__bf16*)alloc((size_t)Opad * Kpad * 2);
    long total = (long)Opad * Kpad;
    k_cvt_pad<<<cdiv(total, 256), 256, 0, stream>>>(w, O, K, dst, Kpad, total);
    return dst;
  };
  for (int i = 0; i < 3; ++i) {
    wbm[2 * i]     = cvt_w(Lm[i].wl, Lm[i].O, Km[i]);
    wbm[2 * i + 1] = cvt_w(Lm[i].wr, Lm[i].O, Km[i]);
    wbc[2 * i]     = cvt_w(Lc[i].wl, Lc[i].O, Kc[i]);
    wbc[2 * i + 1] = cvt_w(Lc[i].wr, Lc[i].O, Kc[i]);
  }

  // ---- run both stacks (sequential, shared scratch) ----
  run_stack(x,  N_MOL, 78, ei,  E_MOL, batch,  Lm, wbm,
            inb, outb, aggf, aggb, sums_m, cnt_m, stream);
  run_stack(xc, N_CLQ, 92, eic, E_CLQ, batchc, Lc, wbc,
            inb, outb, aggf, aggb, sums_c, cnt_c, stream);

  // ---- finalize: mean + concat -> d_out [4096, 256] f32 ----
  k_finalize<<<cdiv((size_t)NUM_GRAPHS * 256, 256), 256, 0, stream>>>(
      sums_m, cnt_m, sums_c, cnt_c, (float*)d_out);
}
